// PesudoLabelGenerator_v1_8_ot_rev_poss_3702261809692
// MI455X (gfx1250) — matrix operand order repaired and checked by reading, not compiled, not run on hardware
//
#include <hip/hip_runtime.h>

#define OC 19  // OUT_CH

typedef __attribute__((ext_vector_type(16))) __bf16       bf16x16;
typedef __attribute__((ext_vector_type(8)))  float        f32x8;
typedef __attribute__((ext_vector_type(4)))  unsigned int u32x4;

static __device__ __forceinline__ unsigned short f2bf(float f) {
  union { float f; unsigned u; } v; v.f = f;
  unsigned u = v.u;
  unsigned r = u + 0x7FFFu + ((u >> 16) & 1u);  // RNE to bf16
  return (unsigned short)(r >> 16);
}

// ---------------------------------------------------------------- init
__global__ void k_init(float* rowsum, float* colsum, unsigned long long* packed,
                       int* winner, int* presence, int* flag,
                       int N1p, int N2p, int N1) {
  int t = blockIdx.x * blockDim.x + threadIdx.x;
  if (t < N1p) { rowsum[t] = 0.0f; winner[t] = -1; }
  if (t < N2p) { colsum[t] = 0.0f; packed[t] = 0xFFFFFFFFull; } // val=0, idx=0
  if (t < N1 + 2) presence[t] = 0;
  if (t == 0) *flag = 0;
}

// ------------------------------------------- diff = inv(posses[1]) @ posses[0]
__global__ void k_diff(const float* posses, float* diff) {
  if (threadIdx.x != 0 || blockIdx.x != 0) return;
  float M[4][8];
  for (int r = 0; r < 4; ++r)
    for (int c = 0; c < 4; ++c) {
      M[r][c]     = posses[16 + r * 4 + c];  // P1
      M[r][4 + c] = (r == c) ? 1.0f : 0.0f;
    }
  for (int p = 0; p < 4; ++p) {
    int piv = p;
    for (int r = p + 1; r < 4; ++r)
      if (fabsf(M[r][p]) > fabsf(M[piv][p])) piv = r;
    if (piv != p)
      for (int c = 0; c < 8; ++c) { float t = M[p][c]; M[p][c] = M[piv][c]; M[piv][c] = t; }
    float inv = 1.0f / M[p][p];
    for (int c = 0; c < 8; ++c) M[p][c] *= inv;
    for (int r = 0; r < 4; ++r)
      if (r != p) {
        float f = M[r][p];
        for (int c = 0; c < 8; ++c) M[r][c] -= f * M[p][c];
      }
  }
  for (int r = 0; r < 4; ++r)
    for (int c = 0; c < 4; ++c) {
      float s = 0.0f;
      for (int k = 0; k < 4; ++k) s += M[r][4 + k] * posses[k * 4 + c];  // inv(P1)@P0
      diff[r * 4 + c] = s;
    }
}

// ------------------------------------------------ per-point prep (sur side)
__global__ void k_prep_sur(const float* __restrict__ F, const float* __restrict__ coords,
                           const float* __restrict__ diff, unsigned short* __restrict__ Bb,
                           float* __restrict__ al, float* __restrict__ yn, float* __restrict__ w2,
                           int N2, int N2p, int C, int Cp) {
  int j = blockIdx.x * blockDim.x + threadIdx.x;
  if (j >= N2p) return;
  if (j < N2) {
    float x = coords[j * 3 + 0], y = coords[j * 3 + 1], z = coords[j * 3 + 2];
    float a0 = diff[0] * x + diff[1] * y + diff[2]  * z + diff[3];
    float a1 = diff[4] * x + diff[5] * y + diff[6]  * z + diff[7];
    float a2 = diff[8] * x + diff[9] * y + diff[10] * z + diff[11];
    al[j * 3 + 0] = a0; al[j * 3 + 1] = a1; al[j * 3 + 2] = a2;
    yn[j] = a0 * a0 + a1 * a1 + a2 * a2;
    float s = 0.0f;
    for (int k = 0; k < C; ++k) {
      float f = F[(size_t)j * C + k];
      s += f * f;
      Bb[(size_t)j * Cp + k] = f2bf(f);
    }
    for (int k = C; k < Cp; ++k) Bb[(size_t)j * Cp + k] = 0;
    w2[j] = sqrtf(s);
  } else {
    for (int k = 0; k < Cp; ++k) Bb[(size_t)j * Cp + k] = 0;
    al[j * 3 + 0] = al[j * 3 + 1] = al[j * 3 + 2] = 0.0f;
    yn[j] = 0.0f; w2[j] = 0.0f;
  }
}

// ------------------------------------------------ per-point prep (mean side)
__global__ void k_prep_mean(const float* __restrict__ MF, const float* __restrict__ ori,
                            unsigned short* __restrict__ Ab,
                            float* __restrict__ xn, float* __restrict__ w1,
                            int N1, int N1p, int C, int Cp) {
  int i = blockIdx.x * blockDim.x + threadIdx.x;
  if (i >= N1p) return;
  if (i < N1) {
    float s = 0.0f;
    for (int k = 0; k < C; ++k) {
      float f = MF[(size_t)i * C + k];
      s += f * f;
      Ab[(size_t)i * Cp + k] = f2bf(f);
    }
    for (int k = C; k < Cp; ++k) Ab[(size_t)i * Cp + k] = 0;
    w1[i] = sqrtf(s);
    float x = ori[i * 3 + 0], y = ori[i * 3 + 1], z = ori[i * 3 + 2];
    xn[i] = x * x + y * y + z * z;
  } else {
    for (int k = 0; k < Cp; ++k) Ab[(size_t)i * Cp + k] = 0;
    w1[i] = 0.0f; xn[i] = 0.0f;
  }
}

// ------------------------------------------- WMMA fragment dot (16x16 tile)
static __device__ __forceinline__ f32x8 wmma_dot(const unsigned short* __restrict__ Ab,
                                                 const unsigned short* __restrict__ Bb,
                                                 int row0, int col0, int Cp, int lane) {
  f32x8 acc = {};
  int m    = lane & 15;
  int aoff = (lane >> 4) * 8;    // A: interleaved K halves per §7.12.2
  int boff = (lane >> 4) * 8;    // B: interleaved K halves, same as A
  const unsigned short* ap = Ab + (size_t)(row0 + m) * Cp;
  const unsigned short* bp = Bb + (size_t)(col0 + m) * Cp;
  for (int k0 = 0; k0 < Cp; k0 += 32) {
    union { bf16x16 v; u32x4 q[2]; } A, B;
    A.q[0] = *(const u32x4*)(ap + k0 + aoff);
    A.q[1] = *(const u32x4*)(ap + k0 + 16 + aoff);
    B.q[0] = *(const u32x4*)(bp + k0 + boff);
    B.q[1] = *(const u32x4*)(bp + k0 + 16 + boff);
    acc = __builtin_amdgcn_wmma_f32_16x16x32_bf16(false, A.v, false, B.v,
                                                  (short)0, acc, false, false);
  }
  return acc;
}

// K(i,j) from the fused epilogue
static __device__ __forceinline__ float k_value(float dot, float w1i, float w2j,
                                                float xni, float ynj, float dotc) {
  float cd  = fmaxf(xni + ynj - 2.0f * dotc, 0.0f) * 4.0f;      // /0.25
  float fdc = 1.0f - dot / fmaxf(w1i * w2j, 1e-8f);
  float dm  = 1.0f + fdc - __expf(-0.5f * cd);
  return (cd < 100.0f) ? __expf(dm * (-1.0f / 0.03f)) : 0.0f;
}

// --------------------------------- pass 1: row sums (K@u1) & col sums (K.T@u2)
__global__ void __launch_bounds__(128)
k_gemm_sums(const unsigned short* __restrict__ Ab, const unsigned short* __restrict__ Bb, int Cp,
            const float* __restrict__ w1, const float* __restrict__ w2,
            const float* __restrict__ xn, const float* __restrict__ yn,
            const float* __restrict__ ori, const float* __restrict__ al,
            float* __restrict__ rowsum, float* __restrict__ colsum, int N1, int N2) {
  int lane = threadIdx.x & 31;
  int wave = threadIdx.x >> 5;
  int row0 = blockIdx.y * 16;
  int col0 = blockIdx.x * 64 + wave * 16;
  f32x8 acc = wmma_dot(Ab, Bb, row0, col0, Cp, lane);

  int half = lane >> 4, n = lane & 15;
  int j = col0 + n;
  bool jok = (j < N2);
  float yj = 0.f, wj = 0.f, ax = 0.f, ay = 0.f, az = 0.f;
  if (jok) { yj = yn[j]; wj = w2[j]; ax = al[j*3]; ay = al[j*3+1]; az = al[j*3+2]; }

  float cs = 0.0f;
  for (int r = 0; r < 8; ++r) {
    int i = row0 + r + 8 * half;
    float Kv = 0.0f;
    if (jok && i < N1) {
      float dotc = ori[i*3] * ax + ori[i*3+1] * ay + ori[i*3+2] * az;
      Kv = k_value(acc[r], w1[i], wj, xn[i], yj, dotc);
    }
    cs += Kv;
    float rs = Kv;
    rs += __shfl_xor(rs, 1, 32);
    rs += __shfl_xor(rs, 2, 32);
    rs += __shfl_xor(rs, 4, 32);
    rs += __shfl_xor(rs, 8, 32);
    if (n == 0 && i < N1) atomicAdd(&rowsum[i], rs);
  }
  cs += __shfl_xor(cs, 16, 32);
  if (half == 0 && jok) atomicAdd(&colsum[j], cs);
}

// ----------------------------------------------- b = u2 / (rowsum*u1 + eps)
__global__ void k_compute_b(const float* rowsum, float* bvec, int N1, int N1p,
                            float u1, float u2) {
  int i = blockIdx.x * blockDim.x + threadIdx.x;
  if (i >= N1p) return;
  bvec[i] = (i < N1) ? u2 / (rowsum[i] * u1 + 1e-16f) : 0.0f;
}

// ---------------- pass 2: per-column argmax_i of b[i]*K(i,j), packed atomicMax
__global__ void __launch_bounds__(128)
k_gemm_argmax(const unsigned short* __restrict__ Ab, const unsigned short* __restrict__ Bb, int Cp,
              const float* __restrict__ w1, const float* __restrict__ w2,
              const float* __restrict__ xn, const float* __restrict__ yn,
              const float* __restrict__ ori, const float* __restrict__ al,
              const float* __restrict__ bvec, unsigned long long* __restrict__ packed,
              int N1, int N2) {
  int lane = threadIdx.x & 31;
  int wave = threadIdx.x >> 5;
  int row0 = blockIdx.y * 16;
  int col0 = blockIdx.x * 64 + wave * 16;
  f32x8 acc = wmma_dot(Ab, Bb, row0, col0, Cp, lane);

  int half = lane >> 4, n = lane & 15;
  int j = col0 + n;
  bool jok = (j < N2);
  float yj = 0.f, wj = 0.f, ax = 0.f, ay = 0.f, az = 0.f;
  if (jok) { yj = yn[j]; wj = w2[j]; ax = al[j*3]; ay = al[j*3+1]; az = al[j*3+2]; }

  float bestv = -1.0f;
  int   besti = 0x7FFFFFFF;
  for (int r = 0; r < 8; ++r) {           // i ascending -> first-index ties kept
    int i = row0 + r + 8 * half;
    if (jok && i < N1) {
      float dotc = ori[i*3] * ax + ori[i*3+1] * ay + ori[i*3+2] * az;
      float v = k_value(acc[r], w1[i], wj, xn[i], yj, dotc) * bvec[i];
      if (v > bestv) { bestv = v; besti = i; }
    }
  }
  float ov = __shfl_xor(bestv, 16, 32);
  int   oi = __shfl_xor(besti, 16, 32);
  if (ov > bestv || (ov == bestv && oi < besti)) { bestv = ov; besti = oi; }
  if (half == 0 && jok && bestv >= 0.0f) {
    unsigned long long pk = ((unsigned long long)__float_as_uint(bestv) << 32)
                          | (unsigned long long)(0xFFFFFFFFu - (unsigned)besti);
    atomicMax(&packed[j], pk);            // val>=0 -> float bits monotonic
  }
}

// ---------------------------------------------------------------- softmax
__global__ void k_softmax(const float* __restrict__ sv, float* __restrict__ outp, int N1) {
  int i = blockIdx.x * blockDim.x + threadIdx.x;
  if (i >= N1) return;
  float m = sv[(size_t)i * OC];
  for (int c = 1; c < OC; ++c) m = fmaxf(m, sv[(size_t)i * OC + c]);
  float e[OC], s = 0.0f;
  for (int c = 0; c < OC; ++c) { e[c] = __expf(sv[(size_t)i * OC + c] - m); s += e[c]; }
  float inv = 1.0f / s;
  for (int c = 0; c < OC; ++c) outp[(size_t)i * OC + c] = e[c] * inv;
}

// ------------------------------------- decode argmax, gather vals, any-flag
__global__ void k_vals(const unsigned long long* __restrict__ packed, const int* __restrict__ gt,
                       const float* __restrict__ svp, int* __restrict__ sm_index,
                       float* __restrict__ vals, int* flag, const int* lgp, int N1, int N2) {
  int j = blockIdx.x * blockDim.x + threadIdx.x;
  int lg = *lgp;
  if (j >= lg || j >= N2) return;
  unsigned low = (unsigned)(packed[j] & 0xFFFFFFFFull);
  int sm = (int)(0xFFFFFFFFu - low);
  if (sm < 0 || sm >= N1) sm = 0;
  sm_index[j] = sm;
  int g = gt[j]; g = g < 0 ? 0 : (g > OC - 1 ? OC - 1 : g);
  float v = svp[(size_t)sm * OC + g];
  vals[j] = v;
  if (v > 0.1f) atomicOr(flag, 1);
}

// ----------------------- mask, scatter winners (max-j dedup), trust presence
__global__ void k_mask(const int* __restrict__ sm_index, const float* __restrict__ vals,
                       const int* flag, int* __restrict__ winner, int* __restrict__ presence,
                       const int* lgp) {
  int j = blockIdx.x * blockDim.x + threadIdx.x;
  int lg = *lgp;
  if (j >= lg) return;
  bool mask = (*flag) ? (vals[j] > 0.1f) : (vals[j] > 0.0f);
  if (mask) {
    int sm = sm_index[j];
    atomicMax(&winner[sm], j);
    presence[sm + 1] = 1;
  } else {
    presence[0] = 1;   // value -1 present in trust
  }
}

__global__ void k_scatter(const int* __restrict__ winner, const int* __restrict__ gt,
                          float* __restrict__ svp, int N1) {
  int i = blockIdx.x * blockDim.x + threadIdx.x;
  if (i >= N1) return;
  int w = winner[i];
  if (w >= 0) {
    int g = gt[w];
    for (int c = 0; c < OC; ++c) svp[(size_t)i * OC + c] = (c == g) ? 1.0f : 0.0f;
  }
}

__global__ void k_pre(const float* __restrict__ svp, float* __restrict__ pre, int N1) {
  int i = blockIdx.x * blockDim.x + threadIdx.x;
  if (i >= N1) return;
  float best = svp[(size_t)i * OC];
  int bi = 0;
  for (int c = 1; c < OC; ++c) {
    float v = svp[(size_t)i * OC + c];
    if (v > best) { best = v; bi = c; }
  }
  pre[i] = (float)bi;
}

// sorted unique of {masked sm_index} ∪ {-1 if any unmasked}, padded with -1
__global__ void k_trust(const int* __restrict__ presence, float* __restrict__ t1,
                        float* __restrict__ t2, const int* lgp, int N1) {
  if (threadIdx.x != 0 || blockIdx.x != 0) return;
  int lg = *lgp;
  int cnt = 0;
  for (int v = -1; v < N1 && cnt < lg; ++v)
    if (presence[v + 1]) { t1[cnt] = (float)v; t2[cnt] = (float)v; ++cnt; }
  for (; cnt < lg; ++cnt) { t1[cnt] = -1.0f; t2[cnt] = -1.0f; }
}

// ================================================================ host glue
extern "C" void kernel_launch(void* const* d_in, const int* in_sizes, int n_in,
                              void* d_out, int out_size, void* d_ws, size_t ws_size,
                              hipStream_t stream) {
  const float* F    = (const float*)d_in[0];  // sur_sv_feature (N2,C)
  const float* sc   = (const float*)d_in[1];  // sur_sv_coords  (N2,3)
  const int*   gt   = (const int*)  d_in[2];  // sur_sv_gt      (N2)
  const float* svp  = (const float*)d_in[3];  // sv_prob        (N1,19)
  const float* mf   = (const float*)d_in[4];  // mean_features  (N1,C)
  const float* oc   = (const float*)d_in[5];  // ori_coords     (N1,3)
  const float* poss = (const float*)d_in[6];  // posses         (2,4,4)
  const int*   lgp  = (const int*)  d_in[8];  // lg (device scalar)
  (void)n_in; (void)ws_size;

  int N2 = in_sizes[2];
  int C  = in_sizes[0] / (N2 > 0 ? N2 : 1);
  int N1 = in_sizes[5] / 3;
  // out_size = 2*lg + N1*C + 3*N1 + N1 + 19*N1
  int lg = (out_size - N1 * (C + 23)) / 2;
  if (lg < 0) lg = 0;
  if (lg > N2) lg = N2;

  int N1p = (N1 + 15) & ~15;
  int N2p = (N2 + 63) & ~63;
  int Cp  = (C + 31) & ~31;

  char* w = (char*)d_ws;
  auto carve = [&](size_t bytes) -> char* {
    char* p = w;
    w += (bytes + 255) & ~(size_t)255;
    return p;
  };
  unsigned short* Ab   = (unsigned short*)carve((size_t)N1p * Cp * 2);
  unsigned short* Bb   = (unsigned short*)carve((size_t)N2p * Cp * 2);
  float* al            = (float*)carve((size_t)N2p * 3 * 4);
  float* yn            = (float*)carve((size_t)N2p * 4);
  float* w2v           = (float*)carve((size_t)N2p * 4);
  float* xn            = (float*)carve((size_t)N1p * 4);
  float* w1v           = (float*)carve((size_t)N1p * 4);
  float* rowsum        = (float*)carve((size_t)N1p * 4);
  float* colsum        = (float*)carve((size_t)N2p * 4);
  float* bvec          = (float*)carve((size_t)N1p * 4);
  unsigned long long* packed = (unsigned long long*)carve((size_t)N2p * 8);
  int*   sm_index      = (int*)carve((size_t)N2p * 4);
  float* vals          = (float*)carve((size_t)N2p * 4);
  int*   winner        = (int*)carve((size_t)N1p * 4);
  int*   presence      = (int*)carve((size_t)(N1 + 2) * 4);
  float* diff          = (float*)carve(64);
  int*   flag          = (int*)carve(4);

  float* out      = (float*)d_out;
  float* o_trust1 = out;
  float* o_mean   = out + lg;
  float* o_ori    = o_mean + (size_t)N1 * C;
  float* o_pre    = o_ori + (size_t)N1 * 3;
  float* o_svp    = o_pre + N1;
  float* o_trust2 = o_svp + (size_t)N1 * OC;

  int mx = N1p > N2p ? N1p : N2p;
  if (N1 + 2 > mx) mx = N1 + 2;

  k_init<<<(mx + 255) / 256, 256, 0, stream>>>(rowsum, colsum, packed, winner,
                                               presence, flag, N1p, N2p, N1);
  k_diff<<<1, 32, 0, stream>>>(poss, diff);
  k_prep_sur <<<(N2p + 255) / 256, 256, 0, stream>>>(F, sc, diff, Bb, al, yn, w2v,
                                                     N2, N2p, C, Cp);
  k_prep_mean<<<(N1p + 255) / 256, 256, 0, stream>>>(mf, oc, Ab, xn, w1v, N1, N1p, C, Cp);

  dim3 g(N2p / 64, N1p / 16);
  k_gemm_sums<<<g, 128, 0, stream>>>(Ab, Bb, Cp, w1v, w2v, xn, yn, oc, al,
                                     rowsum, colsum, N1, N2);
  k_compute_b<<<(N1p + 255) / 256, 256, 0, stream>>>(rowsum, bvec, N1, N1p,
                                                     1.0f / (float)N2, 1.0f / (float)N1);
  k_gemm_argmax<<<g, 128, 0, stream>>>(Ab, Bb, Cp, w1v, w2v, xn, yn, oc, al,
                                       bvec, packed, N1, N2);

  k_softmax<<<(N1 + 255) / 256, 256, 0, stream>>>(svp, o_svp, N1);
  k_vals<<<(N2p + 255) / 256, 256, 0, stream>>>(packed, gt, o_svp, sm_index, vals,
                                                flag, lgp, N1, N2);
  k_mask<<<(N2p + 255) / 256, 256, 0, stream>>>(sm_index, vals, flag, winner,
                                                presence, lgp);
  k_scatter<<<(N1 + 255) / 256, 256, 0, stream>>>(winner, gt, o_svp, N1);
  k_pre<<<(N1 + 255) / 256, 256, 0, stream>>>(o_svp, o_pre, N1);
  k_trust<<<1, 32, 0, stream>>>(presence, o_trust1, o_trust2, lgp, N1);

  hipMemcpyAsync(o_mean, mf, (size_t)N1 * C * sizeof(float),
                 hipMemcpyDeviceToDevice, stream);
  hipMemcpyAsync(o_ori, oc, (size_t)N1 * 3 * sizeof(float),
                 hipMemcpyDeviceToDevice, stream);
}
